// MultiGAT_72318659330634
// MI455X (gfx1250) — matrix operand order repaired and checked by reading, not compiled, not run on hardware
//
#include <hip/hip_runtime.h>
#include <hip/hip_bf16.h>
#include <stdint.h>

// ---------------------------------------------------------------------------
// Two-layer GAT (H=2 heads, C=64) for MI455X / gfx1250, wave32.
//   xw = X @ W^T          -> WMMA f16 (f32 accumulate), vectorized f32 loads
//   a_src/a_dst           -> per-(node,head) dots
//   segment softmax       -> 3 edge passes (atomicMax key / atomicAdd / scatter)
//   bias + ELU            -> elementwise
// ---------------------------------------------------------------------------

typedef __attribute__((ext_vector_type(16))) _Float16 v16h;
typedef __attribute__((ext_vector_type(8)))  float    v8f;

#define HGAT_F 128            // feature width (both layers)
#define HGAT_C 64             // channels per head
#define NEG_SLOPE 0.2f
#define GAT_EPS 1e-16f

// ---------------- monotone float<->uint key for atomic segment-max ---------
__device__ __forceinline__ unsigned enc_key(float f) {
    unsigned u = __float_as_uint(f);
    return (u & 0x80000000u) ? ~u : (u | 0x80000000u);
}
__device__ __forceinline__ float dec_key(unsigned k) {
    unsigned u = (k & 0x80000000u) ? (k & 0x7FFFFFFFu) : ~k;
    return __uint_as_float(u);
}

__device__ __forceinline__ float4 ldg4(const float* p) {
    return *(const float4*)p;
}

// ---------------------------------------------------------------------------
// GEMM: Y[N,128] = X[N,128] @ W^T  with W[128,128] row-major (out x in).
// 256 threads = 8 waves per block. Block owns 16 rows; wave w owns output
// columns [16w, 16w+16). K loop: 4 x v_wmma_f32_16x16x32_f16.
//
// A (16-bit 16x32, MxK) per ISA table: lane L (group g=L/16) holds row
// M=L%16; its 16 half-slots cover exactly two contiguous K runs:
//   slots 0..7  <- K = k0 + g*8      .. +7
//   slots 8..15 <- K = k0 + 16 + g*8 .. +7
// B (16-bit 32x16, KxN): lane L holds col N=L%16; slots 0..15 cover the same
// two K runs as A (k0 + g*8 .. +7, k0 + 16 + g*8 .. +7).
// C/D (f32 16x16): VGPR v, lane L -> row = v + 8*g, col = L%16.
//
// Out-of-range A rows are pointer-clamped (garbage only reaches D rows >= N,
// which are masked at the store) -> no divergence anywhere near the WMMA.
// ---------------------------------------------------------------------------
__global__ void __launch_bounds__(256)
gat_gemm_wmma(const float* __restrict__ X, const float* __restrict__ W,
              float* __restrict__ Y, int N) {
    const int lane = threadIdx.x & 31;
    const int wave = threadIdx.x >> 5;       // 0..7 -> output col tile
    const int r0   = blockIdx.x * 16;
    const int o0   = wave * 16;
    const int m    = lane & 15;
    const int g    = lane >> 4;              // lane group 0/1

    int arow = r0 + m;
    if (arow >= N) arow = N - 1;             // clamp; masked at store
    const float* __restrict__ xr = X + (size_t)arow * HGAT_F + g * 8;
    const float* __restrict__ wr = W + (size_t)(o0 + m) * HGAT_F + g * 8;

    v8f acc = {};
#pragma unroll
    for (int k0 = 0; k0 < HGAT_F; k0 += 32) {
        // ---- A: two contiguous 8-float runs -> 4x float4 ----
        const float4 a0 = ldg4(xr + k0);
        const float4 a1 = ldg4(xr + k0 + 4);
        const float4 a2 = ldg4(xr + k0 + 16);
        const float4 a3 = ldg4(xr + k0 + 20);
        // ---- B: two contiguous 8-float runs (same K order as A) -> 4x float4 ----
        const float4 b0 = ldg4(wr + k0);
        const float4 b1 = ldg4(wr + k0 + 4);
        const float4 b2 = ldg4(wr + k0 + 16);
        const float4 b3 = ldg4(wr + k0 + 20);

        v16h a, b;
        a[0]=(_Float16)a0.x; a[1]=(_Float16)a0.y; a[2]=(_Float16)a0.z; a[3]=(_Float16)a0.w;
        a[4]=(_Float16)a1.x; a[5]=(_Float16)a1.y; a[6]=(_Float16)a1.z; a[7]=(_Float16)a1.w;
        a[8]=(_Float16)a2.x; a[9]=(_Float16)a2.y; a[10]=(_Float16)a2.z; a[11]=(_Float16)a2.w;
        a[12]=(_Float16)a3.x; a[13]=(_Float16)a3.y; a[14]=(_Float16)a3.z; a[15]=(_Float16)a3.w;
        b[0]=(_Float16)b0.x; b[1]=(_Float16)b0.y; b[2]=(_Float16)b0.z; b[3]=(_Float16)b0.w;
        b[4]=(_Float16)b1.x; b[5]=(_Float16)b1.y; b[6]=(_Float16)b1.z; b[7]=(_Float16)b1.w;
        b[8]=(_Float16)b2.x; b[9]=(_Float16)b2.y; b[10]=(_Float16)b2.z; b[11]=(_Float16)b2.w;
        b[12]=(_Float16)b3.x; b[13]=(_Float16)b3.y; b[14]=(_Float16)b3.z; b[15]=(_Float16)b3.w;

        acc = __builtin_amdgcn_wmma_f32_16x16x32_f16(
            /*neg_a=*/false, a, /*neg_b=*/false, b,
            /*c_mod=*/(short)0, acc, /*reuse_a=*/false, /*reuse_b=*/false);
    }

#pragma unroll
    for (int v = 0; v < 8; ++v) {
        const int row = r0 + v + 8 * g;
        if (row < N) Y[(size_t)row * HGAT_F + o0 + m] = acc[v];
    }
}

// ---------------- per-(node,head) attention coefficients -------------------
__global__ void gat_attn(const float* __restrict__ xw,
                         const float* __restrict__ att_s,
                         const float* __restrict__ att_d,
                         float* __restrict__ a_src, float* __restrict__ a_dst,
                         int N) {
    int i = blockIdx.x * blockDim.x + threadIdx.x;   // n*2 + h
    if (i >= N * 2) return;
    const int n = i >> 1, h = i & 1;
    const float* __restrict__ row = xw + (size_t)n * HGAT_F + h * HGAT_C;
    const float* __restrict__ as  = att_s + h * HGAT_C;
    const float* __restrict__ ad  = att_d + h * HGAT_C;
    float s = 0.f, d = 0.f;
#pragma unroll 4
    for (int c = 0; c < HGAT_C; c += 4) {
        const float4 v = ldg4(row + c);
        const float4 vs = ldg4(as + c);
        const float4 vd = ldg4(ad + c);
        s += v.x * vs.x + v.y * vs.y + v.z * vs.z + v.w * vs.w;
        d += v.x * vd.x + v.y * vd.y + v.z * vd.z + v.w * vd.w;
    }
    a_src[i] = s; a_dst[i] = d;
}

// ---------------- edge pass 1: alpha + segment max -------------------------
__global__ void gat_edge_max(const long long* __restrict__ esrc,
                             const long long* __restrict__ edst,
                             int E, int N,
                             const float* __restrict__ a_src,
                             const float* __restrict__ a_dst,
                             float* __restrict__ alpha,
                             unsigned* __restrict__ amax) {
    int e = blockIdx.x * blockDim.x + threadIdx.x;
    const int EE = E + N;
    if (e >= EE) return;
    int s, d;
    if (e < E) { s = (int)esrc[e]; d = (int)edst[e]; } else { s = d = e - E; }
#pragma unroll
    for (int h = 0; h < 2; ++h) {
        float al = a_src[s * 2 + h] + a_dst[d * 2 + h];
        al = (al > 0.f) ? al : NEG_SLOPE * al;       // leaky_relu
        alpha[(size_t)e * 2 + h] = al;
        atomicMax(&amax[d * 2 + h], enc_key(al));
    }
}

// ---------------- edge pass 2: exp + segment sum ---------------------------
__global__ void gat_edge_exp(const long long* __restrict__ edst,
                             int E, int N,
                             float* __restrict__ alpha,      // in: alpha, out: ex
                             const unsigned* __restrict__ amax,
                             float* __restrict__ denom) {
    int e = blockIdx.x * blockDim.x + threadIdx.x;
    const int EE = E + N;
    if (e >= EE) return;
    const int d = (e < E) ? (int)edst[e] : (e - E);
#pragma unroll
    for (int h = 0; h < 2; ++h) {
        const float mx = dec_key(amax[d * 2 + h]);
        const float ex = __expf(alpha[(size_t)e * 2 + h] - mx);
        alpha[(size_t)e * 2 + h] = ex;
        atomicAdd(&denom[d * 2 + h], ex);
    }
}

// ---------------- edge pass 3: weighted scatter (one wave32 per edge) ------
__global__ void __launch_bounds__(256)
gat_scatter(const long long* __restrict__ esrc,
            const long long* __restrict__ edst,
            int E, int N,
            const float* __restrict__ xw,
            const float* __restrict__ ex,
            const float* __restrict__ denom,
            float* __restrict__ acc) {
    const int lane = threadIdx.x & 31;
    const int e    = (blockIdx.x * blockDim.x + threadIdx.x) >> 5;
    const int EE   = E + N;
    if (e >= EE) return;                              // whole wave exits together
    int s, d;
    if (e < E) { s = (int)esrc[e]; d = (int)edst[e]; } else { s = d = e - E; }

    const float c0 = ex[(size_t)e * 2 + 0] / (denom[d * 2 + 0] + GAT_EPS);
    const float c1 = ex[(size_t)e * 2 + 1] / (denom[d * 2 + 1] + GAT_EPS);
    const float coef = (lane < 16) ? c0 : c1;         // channels lane*4..+3, head=lane/16

    const float* __restrict__ srow = xw + (size_t)s * HGAT_F + lane * 4;
    __builtin_prefetch(srow, 0, 1);                   // global_prefetch_b8
    const float4 v = ldg4(srow);

    float* __restrict__ outp = acc + (size_t)d * HGAT_F + lane * 4;
    atomicAdd(outp + 0, v.x * coef);
    atomicAdd(outp + 1, v.y * coef);
    atomicAdd(outp + 2, v.z * coef);
    atomicAdd(outp + 3, v.w * coef);
}

// ---------------- bias + ELU (alpha=1), in place ---------------------------
__global__ void gat_finalize(float* __restrict__ acc,
                             const float* __restrict__ bias, int total) {
    int i = blockIdx.x * blockDim.x + threadIdx.x;
    if (i >= total) return;
    float v = acc[i] + bias[i & (HGAT_F - 1)];
    acc[i] = (v > 0.f) ? v : (__expf(v) - 1.f);
}

// ---------------------------------------------------------------------------
extern "C" void kernel_launch(void* const* d_in, const int* in_sizes, int n_in,
                              void* d_out, int out_size, void* d_ws, size_t ws_size,
                              hipStream_t stream) {
    (void)n_in; (void)out_size; (void)ws_size;

    const float*     x     = (const float*)d_in[0];
    const long long* edges = (const long long*)d_in[1];   // jnp.int64, shape (2,E) flat
    const float*     W0    = (const float*)d_in[2];
    const float*     as0   = (const float*)d_in[3];
    const float*     ad0   = (const float*)d_in[4];
    const float*     b0    = (const float*)d_in[5];
    const float*     W1    = (const float*)d_in[6];
    const float*     as1   = (const float*)d_in[7];
    const float*     ad1   = (const float*)d_in[8];
    const float*     b1    = (const float*)d_in[9];

    const int N  = in_sizes[0] / HGAT_F;
    const int E  = in_sizes[1] / 2;
    const int EE = E + N;
    const long long* esrc = edges;
    const long long* edst = edges + E;

    // -------- workspace carve-up (all 4-byte types) --------
    char* p = (char*)d_ws;
    float*    xw    = (float*)p;    p += (size_t)N * HGAT_F * 4;
    float*    hbuf  = (float*)p;    p += (size_t)N * HGAT_F * 4;
    float*    a_src = (float*)p;    p += (size_t)N * 2 * 4;
    float*    a_dst = (float*)p;    p += (size_t)N * 2 * 4;
    unsigned* amax  = (unsigned*)p; p += (size_t)N * 2 * 4;
    float*    denom = (float*)p;    p += (size_t)N * 2 * 4;
    float*    alpha = (float*)p;    p += (size_t)EE * 2 * 4;

    float* out = (float*)d_out;

    const int nRowBlk  = (N + 15) / 16;
    const int nAttn    = (N * 2 + 255) / 256;
    const int nEdge    = (EE + 255) / 256;
    const int nScatter = ((EE * 32) + 255) / 256;
    const int nFin     = (N * HGAT_F + 255) / 256;

    const float*    Ws[2]   = {W0, W1};
    const float*    ATs[2]  = {as0, as1};
    const float*    ATd[2]  = {ad0, ad1};
    const float*    Bs[2]   = {b0, b1};
    const float*    Xin[2]  = {x, hbuf};
    float*          Acc[2]  = {hbuf, out};

    for (int layer = 0; layer < 2; ++layer) {
        // 1) xw = Xin @ W^T  (WMMA)
        gat_gemm_wmma<<<nRowBlk, 256, 0, stream>>>(Xin[layer], Ws[layer], xw, N);
        // 2) attention coefficients
        gat_attn<<<nAttn, 256, 0, stream>>>(xw, ATs[layer], ATd[layer], a_src, a_dst, N);
        // 3) init segment-max keys, denoms, accumulator
        hipMemsetAsync(amax,       0, (size_t)N * 2 * 4, stream);
        hipMemsetAsync(denom,      0, (size_t)N * 2 * 4, stream);
        hipMemsetAsync(Acc[layer], 0, (size_t)N * HGAT_F * 4, stream);
        // 4) segment softmax + scatter
        gat_edge_max<<<nEdge, 256, 0, stream>>>(esrc, edst, E, N, a_src, a_dst, alpha, amax);
        gat_edge_exp<<<nEdge, 256, 0, stream>>>(edst, E, N, alpha, amax, denom);
        gat_scatter<<<nScatter, 256, 0, stream>>>(esrc, edst, E, N, xw, alpha, denom, Acc[layer]);
        // 5) bias + ELU
        gat_finalize<<<nFin, 256, 0, stream>>>(Acc[layer], Bs[layer], N * HGAT_F);
    }
}
